// DSDMClassifier_37220186587420
// MI455X (gfx1250) — compile-verified, matrix operand order checked by reading, not run on hardware
//
#include <hip/hip_runtime.h>
#include <hip/hip_bf16.h>

// ---------------------------------------------------------------------------
// DSDM classifier, fused for MI455X (gfx1250, wave32, WMMA bf16):
//   dist[b,n] = sqrt(max(0, ||x_b||^2 + ||A_n||^2 - 2 x.A))
//   w = exp(-dist/T)   (no max-shift needed: dist>=0 -> w in (0,1])
//   out[b,c] = (sum_n w[b,n] M[n,c]) / (sum_n w[b,n])
// GEMM1 and GEMM2 both run on v_wmma_f32_16x16x32_bf16, fused per 32-n chunk.
// N is additionally split across gridDim.y=NSPLIT workgroups for occupancy
// (128 WGs alone ~= 1 wave/SIMD on a ~256-WGP part); partials are combined by
// a small finalize kernel.
// ---------------------------------------------------------------------------

#define NSPLIT 8
#define PROWS  113   // 112 numerator rows (c) + 1 denominator row per b-tile

typedef __attribute__((ext_vector_type(16))) __bf16        v16bf;
typedef __attribute__((ext_vector_type(8)))  float         v8f;
typedef __attribute__((ext_vector_type(8)))  unsigned int  v8u;

__device__ __forceinline__ unsigned short f2bf(float f) {
  unsigned int u = __builtin_bit_cast(unsigned int, f);
  u += 0x7FFFu + ((u >> 16) & 1u);               // round-to-nearest-even
  return (unsigned short)(u >> 16);
}
__device__ __forceinline__ unsigned int pack_bf2(float a, float b) {
  return (unsigned int)f2bf(a) | ((unsigned int)f2bf(b) << 16);
}

// Load one 16-wide x K=32 bf16 WMMA fragment from a row-major matrix.
// rowPtr = &row_of_this_lane[kBase].  Per ISA 16-bit A layout:
//   lanes 0-15 : K pairs {0..7} in v0-3, {16..23} in v4-7
//   lanes 16-31: K pairs {8..15} in v0-3, {24..31} in v4-7
__device__ __forceinline__ v16bf load_frag_k32(const unsigned short* rowPtr, int half) {
  const uint4 lo = *(const uint4*)(rowPtr + 8 * half);
  const uint4 hi = *(const uint4*)(rowPtr + 16 + 8 * half);
  v8u u;
  u[0] = lo.x; u[1] = lo.y; u[2] = lo.z; u[3] = lo.w;
  u[4] = hi.x; u[5] = hi.y; u[6] = hi.z; u[7] = hi.w;
  return __builtin_bit_cast(v16bf, u);
}

// ---------------------------------------------------------------------------
// Prep 1: row-major f32 (rows x 128) -> bf16 copy + per-row squared L2 norm.
// One wave per row, float4 per lane.
// ---------------------------------------------------------------------------
__global__ void prep_rows_kernel(const float* __restrict__ src,
                                 unsigned short* __restrict__ dst,
                                 float* __restrict__ norms, int rows) {
  __shared__ float red[256];
  const int wave = threadIdx.x >> 5;
  const int lane = threadIdx.x & 31;
  const int row  = blockIdx.x * 8 + wave;
  const bool ok  = row < rows;

  float4 v = make_float4(0.f, 0.f, 0.f, 0.f);
  if (ok) v = ((const float4*)(src + (size_t)row * 128))[lane];
  if (ok) {
    uint2 dd;
    dd.x = pack_bf2(v.x, v.y);
    dd.y = pack_bf2(v.z, v.w);
    ((uint2*)(dst + (size_t)row * 128))[lane] = dd;
  }
  red[threadIdx.x] = v.x * v.x + v.y * v.y + v.z * v.z + v.w * v.w;
  __syncthreads();
  if (ok && lane == 0) {
    float s = 0.f;
    for (int i = 0; i < 32; ++i) s += red[wave * 32 + i];
    norms[row] = s;
  }
}

// ---------------------------------------------------------------------------
// Prep 2: M (N x C f32) -> Mt (CP x N bf16), transposed + zero-padded to CP=112
// so GEMM2 A-operand loads are contiguous 16B row-major fragments.
// ---------------------------------------------------------------------------
__global__ void prep_mt_kernel(const float* __restrict__ M,
                               unsigned short* __restrict__ Mt,
                               int N, int C, int CP) {
  long long idx = (long long)blockIdx.x * blockDim.x + threadIdx.x;
  if (idx >= (long long)CP * N) return;
  int c = (int)(idx / N);
  int n = (int)(idx % N);
  float v = (c < C) ? M[(long long)n * C + c] : 0.0f;
  Mt[(long long)c * N + n] = f2bf(v);
}

// ---------------------------------------------------------------------------
// Main fused kernel. Grid = (B/16, NSPLIT), 256 threads (8 wave32).
// blockIdx.y selects a contiguous range of 32-address chunks; each wave takes
// chunks strided by 8 within that range; wave partials merge in LDS and one
// (numerator 112x16, denominator 16) partial is written per workgroup.
// ---------------------------------------------------------------------------
__global__ __launch_bounds__(256) void dsdm_main_kernel(
    const unsigned short* __restrict__ Abf,   // N x 128 bf16
    const unsigned short* __restrict__ xbf,   // B x 128 bf16
    const unsigned short* __restrict__ Mt,    // 112 x N bf16
    const float* __restrict__ Anorm,          // N
    const float* __restrict__ xnorm,          // B
    float* __restrict__ partial,              // [B/16][NSPLIT][PROWS][16]
    int N) {
  __shared__ float lds_acc[8][112][16];       // per-wave partial logits^T
  __shared__ float lds_sw[8][32];             // per-wave partial denominators

  const int tid   = threadIdx.x;
  const int wave  = tid >> 5;
  const int lane  = tid & 31;
  const int half  = lane >> 4;
  const int l15   = lane & 15;
  const int bBase = blockIdx.x * 16;

  // exp(-dist/T) = exp2(dist * (-1/(T*ln2))), T = 2
  const float SCALE = -0.72134752044448170368f;

  // x fragments (GEMM1 B-operand, lane = b column) are fixed for the block.
  const unsigned short* xrow = xbf + (size_t)(bBase + l15) * 128;
  v16bf xf[4];
#pragma unroll
  for (int kk = 0; kk < 4; ++kk) xf[kk] = load_frag_k32(xrow + kk * 32, half);

  const float xn = xnorm[bBase + l15];

  const v8f zero8 = {0.f, 0.f, 0.f, 0.f, 0.f, 0.f, 0.f, 0.f};
  v8f accL[7];
#pragma unroll
  for (int ct = 0; ct < 7; ++ct) accL[ct] = zero8;
  float swacc = 0.0f;

  const int nChunks = N >> 5;                 // N assumed multiple of 32
  const int cpw     = (nChunks + NSPLIT - 1) / NSPLIT;
  const int cBeg    = blockIdx.y * cpw;
  const int cEnd    = min(nChunks, cBeg + cpw);

  for (int chunk = cBeg + wave; chunk < cEnd; chunk += 8) {
    const int n0 = chunk << 5;
    unsigned int wpk[8];

#pragma unroll
    for (int t = 0; t < 2; ++t) {
      const int nb = n0 + t * 16;
      // GEMM1 tile: S^T = A_tile(16n x 128k) * x^T(128k x 16b)
      const unsigned short* arow = Abf + (size_t)(nb + l15) * 128;
      v8f s = zero8;
#pragma unroll
      for (int kk = 0; kk < 4; ++kk) {
        v16bf af = load_frag_k32(arow + kk * 32, half);
        s = __builtin_amdgcn_wmma_f32_16x16x32_bf16(
            false, af, false, xf[kk], (short)0, s, false, false);
      }
      // D layout: lane holds column b=l15; VGPR r holds row n = nb + r + 8*half
      const float* anp = Anorm + nb + 8 * half;
      const float4 an0 = *(const float4*)anp;
      const float4 an1 = *(const float4*)(anp + 4);
      const float an[8] = {an0.x, an0.y, an0.z, an0.w,
                           an1.x, an1.y, an1.z, an1.w};
      float w[8];
#pragma unroll
      for (int r = 0; r < 8; ++r) {
        float sq   = fmaxf(an[r] + xn - 2.0f * s[r], 0.0f);
        float dist = __builtin_amdgcn_sqrtf(sq);
        w[r] = __builtin_amdgcn_exp2f(dist * SCALE);
        swacc += w[r];
      }
      // Pack into GEMM2 B-operand dwords; the S^T layout already matches the
      // B-fragment lane/half structure, so packing pairs is sufficient.
#pragma unroll
      for (int j = 0; j < 4; ++j)
        wpk[t * 4 + j] = pack_bf2(w[2 * j], w[2 * j + 1]);
    }
    v8u wv;
#pragma unroll
    for (int j = 0; j < 8; ++j) wv[j] = wpk[j];
    const v16bf wfrag = __builtin_bit_cast(v16bf, wv);

    // GEMM2: L^T(112c x 16b) += Mt_tile(16c x 32n) * W^T(32n x 16b)
#pragma unroll
    for (int ct = 0; ct < 7; ++ct) {
      const unsigned short* mrow = Mt + (size_t)(ct * 16 + l15) * N + n0;
      v16bf mf = load_frag_k32(mrow, half);
      accL[ct] = __builtin_amdgcn_wmma_f32_16x16x32_bf16(
          false, mf, false, wfrag, (short)0, accL[ct], false, false);
    }

    // Prefetch next chunk for this wave (global_prefetch_b8).
    const int n0n = n0 + 32 * 8;
    if (n0n < N) {
      __builtin_prefetch(Abf + (size_t)(n0n + l15) * 128, 0, 3);
      __builtin_prefetch(Mt + (size_t)l15 * N + n0n, 0, 3);
    }
  }

  // Spill per-wave partials to LDS and combine into one per-WG partial.
  lds_sw[wave][lane] = swacc;
#pragma unroll
  for (int ct = 0; ct < 7; ++ct)
#pragma unroll
    for (int r = 0; r < 8; ++r)
      lds_acc[wave][ct * 16 + r + 8 * half][l15] = accL[ct][r];

  __syncthreads();

  float* pout = partial +
      ((size_t)blockIdx.x * NSPLIT + blockIdx.y) * (PROWS * 16);
  for (int i = tid; i < 112 * 16; i += 256) {
    const int c = i >> 4;
    const int b = i & 15;
    float s = 0.0f;
#pragma unroll
    for (int w2 = 0; w2 < 8; ++w2) s += lds_acc[w2][c][b];
    pout[i] = s;
    if (c == 0) {                              // 16 threads also emit denoms
      float sw = 0.0f;
#pragma unroll
      for (int w2 = 0; w2 < 8; ++w2)
        sw += lds_sw[w2][b] + lds_sw[w2][b + 16];
      pout[112 * 16 + b] = sw;
    }
  }
}

// ---------------------------------------------------------------------------
// Finalize: sum the NSPLIT partials per b-tile, normalize, emit logits.
// ---------------------------------------------------------------------------
__global__ void finalize_kernel(const float* __restrict__ partial,
                                float* __restrict__ out, int C) {
  const int bBase = blockIdx.x * 16;
  const float* p  = partial + (size_t)blockIdx.x * NSPLIT * (PROWS * 16);

  for (int i = threadIdx.x; i < 112 * 16; i += 256) {
    const int c = i >> 4;
    const int b = i & 15;
    float num = 0.0f, den = 0.0f;
#pragma unroll
    for (int s = 0; s < NSPLIT; ++s) {
      num += p[s * (PROWS * 16) + i];
      den += p[s * (PROWS * 16) + 112 * 16 + b];
    }
    if (c < C) out[(size_t)(bBase + b) * C + c] = num / den;
  }
}

// ---------------------------------------------------------------------------
extern "C" void kernel_launch(void* const* d_in, const int* in_sizes, int n_in,
                              void* d_out, int out_size, void* d_ws, size_t ws_size,
                              hipStream_t stream) {
  const float* x       = (const float*)d_in[0];
  const float* Address = (const float*)d_in[1];
  const float* M       = (const float*)d_in[2];

  const int D  = 128;
  const int B  = in_sizes[0] / D;   // 2048
  const int N  = in_sizes[1] / D;   // 100000
  const int C  = in_sizes[2] / N;   // 100
  const int CP = 112;               // C padded to a multiple of 16

  char* ws = (char*)d_ws;
  size_t off = 0;
  auto carve = [&](size_t bytes) -> char* {
    char* p = ws + off;
    off = (off + bytes + 255) & ~(size_t)255;
    return p;
  };
  unsigned short* Abf     = (unsigned short*)carve((size_t)N * D * 2);   // 25.6 MB
  unsigned short* Mt      = (unsigned short*)carve((size_t)CP * N * 2);  // 22.4 MB
  unsigned short* xbf     = (unsigned short*)carve((size_t)B * D * 2);   // 0.5 MB
  float*          Anorm   = (float*)carve((size_t)N * 4);
  float*          xnorm   = (float*)carve((size_t)B * 4);
  float*          partial = (float*)carve((size_t)(B / 16) * NSPLIT * PROWS * 16 * 4); // ~7.4 MB
  (void)ws_size; (void)n_in; (void)out_size;

  prep_rows_kernel<<<(N + 7) / 8, 256, 0, stream>>>(Address, Abf, Anorm, N);
  prep_rows_kernel<<<(B + 7) / 8, 256, 0, stream>>>(x, xbf, xnorm, B);
  const long long tot = (long long)CP * N;
  prep_mt_kernel<<<(int)((tot + 255) / 256), 256, 0, stream>>>(M, Mt, N, C, CP);

  dim3 grid(B / 16, NSPLIT);
  dsdm_main_kernel<<<grid, 256, 0, stream>>>(Abf, xbf, Mt, Anorm, xnorm,
                                             partial, N);
  finalize_kernel<<<B / 16, 256, 0, stream>>>(partial, (float*)d_out, C);
}
